// SAGE_53188874994047
// MI455X (gfx1250) — matrix operand order repaired and checked by reading, not compiled, not run on hardware
//
#include <hip/hip_runtime.h>
#include <hip/hip_bf16.h>
#include <stdint.h>

#define NN    20000
#define DD    16
#define FEAT  128
#define GATES 512

typedef __attribute__((ext_vector_type(16))) _Float16 v16h;
typedef __attribute__((ext_vector_type(8)))  float    v8f;

union Frag16 { v16h h; uint4 q[2]; };

__device__ __forceinline__ v8f wmma_f16(v16h a, v16h b, v8f c) {
  // D = A(16x32 f16) * B(32x16 f16) + C(16x16 f32)
  return __builtin_amdgcn_wmma_f32_16x16x32_f16(false, a, false, b, (short)0, c,
                                                false, false);
}

// B (KxN = 32x16) fragment. W is row-major [out][128] == B^T, so B column j is
// W row j (contiguous). Col n0+r; halves 0-7: K = kb + half_id*8 ..; halves 8-15: K = kb+16 + half_id*8 ..
__device__ __forceinline__ v16h load_b_frag(const _Float16* __restrict__ W,
                                            int n0, int kb, int r, int half_id) {
  const uint4* p = reinterpret_cast<const uint4*>(
      W + (size_t)(n0 + r) * FEAT + kb + half_id * 8);
  Frag16 u; u.q[0] = p[0]; u.q[1] = p[2];
  return u.h;
}

// A (MxK = 16x32) fragment from a row-major [16][128] LDS tile.
// Lane row M = r. halves 0-7: K = kb + half_id*8 ..; halves 8-15: K = kb+16 + half_id*8 ..
__device__ __forceinline__ v16h load_a_frag(const _Float16* __restrict__ tile,
                                            int kb, int r, int half_id) {
  Frag16 u;
  u.q[0] = *reinterpret_cast<const uint4*>(tile + r * FEAT + kb + half_id * 8);
  u.q[1] = *reinterpret_cast<const uint4*>(tile + r * FEAT + kb + 16 + half_id * 8);
  return u.h;
}

__device__ __forceinline__ float sigm(float x)  { return 1.f / (1.f + __expf(-x)); }
__device__ __forceinline__ float tanhf_(float x){ return 1.f - 2.f / (__expf(2.f * x) + 1.f); }

// One SAGE-LSTM layer. Block = 256 threads = 8 waves, one 16-node tile per block.
// Wave w owns hidden columns [w*16, w*16+16) of gates i,f,g,o (register-resident B frags).
template<bool FUSE_OUT>
__global__ void __launch_bounds__(256, 1)
sage_lstm_layer(const _Float16* __restrict__ feat,     // [NN][128] f16 node features
                const int*      __restrict__ nbr,      // [NN][16]
                const _Float16* __restrict__ WihH,     // [512][128] f16
                const _Float16* __restrict__ WhhH,     // [512][128] f16
                const float*    __restrict__ bcomb,    // [512] bih+bhh
                const _Float16* __restrict__ WselfH,   // [128][128] f16 (FUSE_OUT only)
                const _Float16* __restrict__ WneighH,  // [128][128] f16 (FUSE_OUT only)
                const float*    __restrict__ bneigh,   // [128]       (FUSE_OUT only)
                _Float16*       __restrict__ outH)     // FUSE: relu(self+neigh) ; else hN
{
  __shared__ __align__(16) _Float16 sXt[16][FEAT];       // gathered neighbor step input
  __shared__ __align__(16) _Float16 sH[2][16][FEAT];     // double-buffered hidden state

  const int tid      = threadIdx.x;
  const int wave     = tid >> 5;
  const int lane     = tid & 31;
  const int half_id  = lane >> 4;
  const int r        = lane & 15;
  const int hbase    = wave * 16;
  const int node_base = blockIdx.x * 16;
  const int grow = tid >> 4;     // cooperative gather: row 0..15
  const int gseg = tid & 15;     // 8-half (16B) segment 0..15

  // h_0 = 0 (2048 halfs = 256 uint4, one per thread)
  reinterpret_cast<uint4*>(&sH[0][0][0])[tid] = make_uint4(0u, 0u, 0u, 0u);

  // Register-resident recurrent weights: 4 gates x 4 K-tiles x 2 matrices.
  v16h Bih[4][4], Bhh[4][4];
  #pragma unroll
  for (int g = 0; g < 4; ++g)
    #pragma unroll
    for (int kt = 0; kt < 4; ++kt) {
      Bih[g][kt] = load_b_frag(WihH, g * FEAT + hbase, kt * 32, r, half_id);
      Bhh[g][kt] = load_b_frag(WhhH, g * FEAT + hbase, kt * 32, r, half_id);
    }

  float bg[4];
  #pragma unroll
  for (int g = 0; g < 4; ++g) bg[g] = bcomb[g * FEAT + hbase + r];

  float cst[8];
  #pragma unroll
  for (int e = 0; e < 8; ++e) cst[e] = 0.f;

  for (int t = 0; t < DD; ++t) {
    const int cur = t & 1, nxt = cur ^ 1;
    __syncthreads();  // sXt free; previous-step h stores visible
    {
      int nb = nbr[(size_t)(node_base + grow) * DD + t];
      *reinterpret_cast<uint4*>(&sXt[grow][gseg * 8]) =
          *reinterpret_cast<const uint4*>(feat + (size_t)nb * FEAT + gseg * 8);
    }
    __syncthreads();  // xt ready, h[cur] ready

    v8f acc[4];
    #pragma unroll
    for (int g = 0; g < 4; ++g)
      #pragma unroll
      for (int e = 0; e < 8; ++e) acc[g][e] = bg[g];

    // Issue all A-fragment LDS loads up front so the 32 WMMAs run as one burst
    // behind a single ds wait instead of 8 interleaved stalls.
    v16h aX[4], aH[4];
    #pragma unroll
    for (int kt = 0; kt < 4; ++kt) {
      aX[kt] = load_a_frag(&sXt[0][0],     kt * 32, r, half_id);
      aH[kt] = load_a_frag(&sH[cur][0][0], kt * 32, r, half_id);
    }

    #pragma unroll
    for (int kt = 0; kt < 4; ++kt)
      #pragma unroll
      for (int g = 0; g < 4; ++g) {
        acc[g] = wmma_f16(aX[kt], Bih[g][kt], acc[g]);
        acc[g] = wmma_f16(aH[kt], Bhh[g][kt], acc[g]);
      }

    // LSTM pointwise update; C/D layout: element e -> (M = e + half_id*8, N = r)
    #pragma unroll
    for (int e = 0; e < 8; ++e) {
      float iv = sigm(acc[0][e]);
      float fv = sigm(acc[1][e]);
      float gv = tanhf_(acc[2][e]);
      float ov = sigm(acc[3][e]);
      float cv = fv * cst[e] + iv * gv;
      cst[e] = cv;
      float hv = ov * tanhf_(cv);
      sH[nxt][e + half_id * 8][hbase + r] = (_Float16)hv;
    }
  }
  __syncthreads();  // DD even -> final h is in buffer 0

  if constexpr (FUSE_OUT) {
    // out = relu(x @ Wself^T + hN @ Wneigh^T + bneigh), stored f16
    *reinterpret_cast<uint4*>(&sXt[grow][gseg * 8]) =
        *reinterpret_cast<const uint4*>(feat + (size_t)(node_base + grow) * FEAT + gseg * 8);
    __syncthreads();

    v8f acc;
    float bo = bneigh[hbase + r];
    #pragma unroll
    for (int e = 0; e < 8; ++e) acc[e] = bo;

    v16h aS[4], aN[4], bS[4], bN[4];
    #pragma unroll
    for (int kt = 0; kt < 4; ++kt) {
      aS[kt] = load_a_frag(&sXt[0][0],   kt * 32, r, half_id);
      aN[kt] = load_a_frag(&sH[0][0][0], kt * 32, r, half_id);
      bS[kt] = load_b_frag(WselfH,  hbase, kt * 32, r, half_id);
      bN[kt] = load_b_frag(WneighH, hbase, kt * 32, r, half_id);
    }
    #pragma unroll
    for (int kt = 0; kt < 4; ++kt) {
      acc = wmma_f16(aS[kt], bS[kt], acc);
      acc = wmma_f16(aN[kt], bN[kt], acc);
    }
    #pragma unroll
    for (int e = 0; e < 8; ++e) {
      float v = fmaxf(acc[e], 0.f);
      int M = e + half_id * 8;
      outH[(size_t)(node_base + M) * FEAT + hbase + r] = (_Float16)v;
    }
  } else {
    // write final hN tile
    *reinterpret_cast<uint4*>(outH + (size_t)(node_base + grow) * FEAT + gseg * 8) =
        *reinterpret_cast<const uint4*>(&sH[0][grow][gseg * 8]);
  }
}

__global__ void cvt_f32_f16(const float* __restrict__ src, _Float16* __restrict__ dst, int n) {
  int i = blockIdx.x * blockDim.x + threadIdx.x;
  if (i < n) dst[i] = (_Float16)src[i];
}

__global__ void vec_add(const float* __restrict__ a, const float* __restrict__ b,
                        float* __restrict__ dst, int n) {
  int i = blockIdx.x * blockDim.x + threadIdx.x;
  if (i < n) dst[i] = a[i] + b[i];
}

// out[n] = sigmoid(h1[n]·Wself2 + hN[n]·Wneigh2 + bneigh2); one wave per node.
__global__ void final_out(const _Float16* __restrict__ h1, const _Float16* __restrict__ hN,
                          const float* __restrict__ Wself2, const float* __restrict__ Wneigh2,
                          const float* __restrict__ bneigh2, float* __restrict__ out) {
  int wave = threadIdx.x >> 5, lane = threadIdx.x & 31;
  int n = blockIdx.x * 8 + wave;
  if (n >= NN) return;
  float s = 0.f;
  #pragma unroll
  for (int j = 0; j < 4; ++j) {
    int k = lane * 4 + j;
    s += (float)h1[(size_t)n * FEAT + k] * Wself2[k]
       + (float)hN[(size_t)n * FEAT + k] * Wneigh2[k];
  }
  #pragma unroll
  for (int off = 16; off > 0; off >>= 1) s += __shfl_xor(s, off, 32);
  if (lane == 0) out[n] = 1.f / (1.f + __expf(-(s + bneigh2[0])));
}

extern "C" void kernel_launch(void* const* d_in, const int* in_sizes, int n_in,
                              void* d_out, int out_size, void* d_ws, size_t ws_size,
                              hipStream_t stream) {
  const float* x       = (const float*)d_in[0];
  const int*   nbr     = (const int*)  d_in[1];
  const float* Wih1    = (const float*)d_in[2];
  const float* Whh1    = (const float*)d_in[3];
  const float* bih1    = (const float*)d_in[4];
  const float* bhh1    = (const float*)d_in[5];
  const float* Wself1  = (const float*)d_in[6];
  const float* Wneigh1 = (const float*)d_in[7];
  const float* bneigh1 = (const float*)d_in[8];
  const float* Wih2    = (const float*)d_in[9];
  const float* Whh2    = (const float*)d_in[10];
  const float* bih2    = (const float*)d_in[11];
  const float* bhh2    = (const float*)d_in[12];
  const float* Wself2  = (const float*)d_in[13];
  const float* Wneigh2 = (const float*)d_in[14];
  const float* bneigh2 = (const float*)d_in[15];
  float* out = (float*)d_out;

  char* ws = (char*)d_ws;
  size_t off = 0;
  auto walloc = [&](size_t bytes) -> void* {
    void* p = ws + off; off += (bytes + 255) & ~(size_t)255; return p;
  };
  _Float16* xh       = (_Float16*)walloc((size_t)NN * FEAT * 2);
  _Float16* h1       = (_Float16*)walloc((size_t)NN * FEAT * 2);
  _Float16* hN2      = (_Float16*)walloc((size_t)NN * FEAT * 2);
  _Float16* Wih1h    = (_Float16*)walloc((size_t)GATES * FEAT * 2);
  _Float16* Whh1h    = (_Float16*)walloc((size_t)GATES * FEAT * 2);
  _Float16* Wih2h    = (_Float16*)walloc((size_t)GATES * FEAT * 2);
  _Float16* Whh2h    = (_Float16*)walloc((size_t)GATES * FEAT * 2);
  _Float16* Wself1h  = (_Float16*)walloc((size_t)FEAT * FEAT * 2);
  _Float16* Wneigh1h = (_Float16*)walloc((size_t)FEAT * FEAT * 2);
  float*    b1       = (float*)walloc((size_t)GATES * 4);
  float*    b2       = (float*)walloc((size_t)GATES * 4);

  const int T = 256;
  cvt_f32_f16<<<(NN * FEAT + T - 1) / T, T, 0, stream>>>(x, xh, NN * FEAT);
  cvt_f32_f16<<<(GATES * FEAT + T - 1) / T, T, 0, stream>>>(Wih1, Wih1h, GATES * FEAT);
  cvt_f32_f16<<<(GATES * FEAT + T - 1) / T, T, 0, stream>>>(Whh1, Whh1h, GATES * FEAT);
  cvt_f32_f16<<<(GATES * FEAT + T - 1) / T, T, 0, stream>>>(Wih2, Wih2h, GATES * FEAT);
  cvt_f32_f16<<<(GATES * FEAT + T - 1) / T, T, 0, stream>>>(Whh2, Whh2h, GATES * FEAT);
  cvt_f32_f16<<<(FEAT * FEAT + T - 1) / T, T, 0, stream>>>(Wself1, Wself1h, FEAT * FEAT);
  cvt_f32_f16<<<(FEAT * FEAT + T - 1) / T, T, 0, stream>>>(Wneigh1, Wneigh1h, FEAT * FEAT);
  vec_add<<<(GATES + T - 1) / T, T, 0, stream>>>(bih1, bhh1, b1, GATES);
  vec_add<<<(GATES + T - 1) / T, T, 0, stream>>>(bih2, bhh2, b2, GATES);

  sage_lstm_layer<true><<<NN / 16, 256, 0, stream>>>(
      xh, nbr, Wih1h, Whh1h, b1, Wself1h, Wneigh1h, bneigh1, h1);
  sage_lstm_layer<false><<<NN / 16, 256, 0, stream>>>(
      h1, nbr, Wih2h, Whh2h, b2, nullptr, nullptr, nullptr, hN2);
  final_out<<<(NN + 7) / 8, 256, 0, stream>>>(h1, hN2, Wself2, Wneigh2, bneigh2, out);

  (void)in_sizes; (void)n_in; (void)out_size; (void)ws_size;
}
